// RNN_15917148799232
// MI455X (gfx1250) — matrix operand order repaired and checked by reading, not compiled, not run on hardware
//
#include <hip/hip_runtime.h>

// ---------------------------------------------------------------------------
// 2-layer tanh RNN on MI455X (gfx1250, wave32, WMMA).
// bf16 operands + f32 accumulation via v_wmma_f32_16x16x32_bf16.
//  - X[t,b,:] = emb[x_tb] @ u0_w^T + u0_b + w0_b hoisted out of the time loop.
//  - Per step: h0n = tanh(h0 @ w0^T + X[t]);
//              h1n = tanh(h1 @ w1^T + h0n @ u1^T + w1_b + u1_b);
//              logits[t] = h1n @ v_w^T + v_b.
// Big GEMMs (X precompute, logits) use 32x64 wave tiles (8 WMMA / 12 b128);
// the small latency-bound recurrent GEMMs use 32x32 wave tiles for more WGs.
// ---------------------------------------------------------------------------

#define VOCAB 10000
#define EMBD  1024
#define HID   1024
#define SEQL  64
#define BATCH 128

typedef __attribute__((ext_vector_type(16))) __bf16 v16bf;
typedef __attribute__((ext_vector_type(8)))  float  v8f;

union FragU { v16bf v; uint4 q[2]; };

__device__ __forceinline__ unsigned short f32_to_bf16(float f) {
  unsigned int u = __float_as_uint(f);
  u += 0x7FFFu + ((u >> 16) & 1u);   // round-to-nearest-even
  return (unsigned short)(u >> 16);
}
__device__ __forceinline__ float bf16_to_f32(unsigned short h) {
  return __uint_as_float(((unsigned int)h) << 16);
}

// A-fragment (16x32 bf16, M = lane&15):
//  lanes 0-15: VGPR0-3 = K 0..7, VGPR4-7 = K 16..23
//  lanes 16-31: VGPR0-3 = K 8..15, VGPR4-7 = K 24..31
__device__ __forceinline__ v16bf load_a_frag(const unsigned short* rowp, int k0, int lh) {
  FragU f;
  f.q[0] = *reinterpret_cast<const uint4*>(rowp + k0 + lh * 8);
  f.q[1] = *reinterpret_cast<const uint4*>(rowp + k0 + 16 + lh * 8);
  return f.v;
}
// B-fragment (32x16 bf16, N = lane&15): lanes 0-15 hold K 0..7,16..23, lanes 16-31 K 8..15,24..31.
// We compute h @ W^T, so B[k][n] = W[n][k] -> K contiguous in memory per column.
__device__ __forceinline__ v16bf load_b_frag(const unsigned short* colp, int k0, int lh) {
  FragU f;
  f.q[0] = *reinterpret_cast<const uint4*>(colp + k0 + lh * 8);
  f.q[1] = *reinterpret_cast<const uint4*>(colp + k0 + 16 + lh * 8);
  return f.v;
}

__device__ __forceinline__ v8f wmma_bf16(v16bf a, v16bf b, v8f c) {
  // (neg_a, A, neg_b, B, c_mod, C, reuse_a, reuse_b)
  return __builtin_amdgcn_wmma_f32_16x16x32_bf16(false, a, false, b, (short)0, c, false, false);
}

// ----------------------------- conversion kernels --------------------------

__global__ void __launch_bounds__(256) k_f32_to_bf16(const float* __restrict__ in,
                                                     unsigned short* __restrict__ out, int n) {
  int i = blockIdx.x * 256 + threadIdx.x;
  if (i < n) out[i] = f32_to_bf16(in[i]);
}

__global__ void __launch_bounds__(256) k_bf16_to_f32(const unsigned short* __restrict__ in,
                                                     float* __restrict__ out, int n) {
  int i = blockIdx.x * 256 + threadIdx.x;
  if (i < n) out[i] = bf16_to_f32(in[i]);
}

// --------------------- X = gather(emb) @ u0^T + u0_b + w0_b -----------------
// M = SEQL*BATCH = 8192, N = HID, K = EMBD.
// Wave tile 32x64 (2 A-frags x 4 B-frags), WG = 8 waves (2M x 4N) -> 64x256.

__global__ void __launch_bounds__(256) k_x_gemm(const int* __restrict__ idx,
                                                const unsigned short* __restrict__ emb,
                                                const unsigned short* __restrict__ u0w,
                                                const float* __restrict__ u0b,
                                                const float* __restrict__ w0b,
                                                float* __restrict__ X) {
  const int wave = threadIdx.x >> 5, lane = threadIdx.x & 31;
  const int lr = lane & 15, lh = lane >> 4;
  const int m0 = blockIdx.x * 64 + (wave & 1) * 32;
  const int n0 = blockIdx.y * 256 + (wave >> 1) * 64;

  const unsigned short* ap[2];
  ap[0] = emb + (long)idx[m0 + lr] * EMBD;
  ap[1] = emb + (long)idx[m0 + 16 + lr] * EMBD;
  const unsigned short* bp[4];
#pragma unroll
  for (int ni = 0; ni < 4; ++ni) bp[ni] = u0w + (long)(n0 + ni * 16 + lr) * EMBD;

  v8f acc[2][4] = {};
  for (int k0 = 0; k0 < EMBD; k0 += 32) {
    v16bf a[2], b[4];
#pragma unroll
    for (int mi = 0; mi < 2; ++mi) a[mi] = load_a_frag(ap[mi], k0, lh);
#pragma unroll
    for (int ni = 0; ni < 4; ++ni) b[ni] = load_b_frag(bp[ni], k0, lh);
#pragma unroll
    for (int mi = 0; mi < 2; ++mi)
#pragma unroll
      for (int ni = 0; ni < 4; ++ni) acc[mi][ni] = wmma_bf16(a[mi], b[ni], acc[mi][ni]);
  }
#pragma unroll
  for (int mi = 0; mi < 2; ++mi)
#pragma unroll
    for (int ni = 0; ni < 4; ++ni) {
      const int col = n0 + ni * 16 + lr;
      const float bias = u0b[col] + w0b[col];
#pragma unroll
      for (int v = 0; v < 8; ++v) {
        const int row = m0 + mi * 16 + lh * 8 + v;  // f32 C/D: M = vgpr + 8*lanehalf
        X[(long)row * HID + col] = acc[mi][ni][v] + bias;
      }
    }
}

// ------------------- hN = tanh(A1@W1^T [+ A2@W2^T] + add + b) ---------------
// M = BATCH, N = HID, K = HID (x2 if A2). Small sequential GEMM: keep 32x32
// wave tiles / 64x128 WG tiles to maximize in-flight workgroups.

__global__ void __launch_bounds__(256) k_rnn_layer(const unsigned short* __restrict__ A1,
                                                   const unsigned short* __restrict__ W1,
                                                   const unsigned short* __restrict__ A2,
                                                   const unsigned short* __restrict__ W2,
                                                   const float* __restrict__ addMat,
                                                   const float* __restrict__ bias1,
                                                   const float* __restrict__ bias2,
                                                   unsigned short* __restrict__ outB) {
  const int wave = threadIdx.x >> 5, lane = threadIdx.x & 31;
  const int lr = lane & 15, lh = lane >> 4;
  const int m0 = blockIdx.x * 64 + (wave & 1) * 32;
  const int n0 = blockIdx.y * 128 + (wave >> 1) * 32;

  v8f acc[2][2] = {};
  {
    const unsigned short* a0p = A1 + (long)(m0 + lr) * HID;
    const unsigned short* a1p = A1 + (long)(m0 + 16 + lr) * HID;
    const unsigned short* b0p = W1 + (long)(n0 + lr) * HID;
    const unsigned short* b1p = W1 + (long)(n0 + 16 + lr) * HID;
    for (int k0 = 0; k0 < HID; k0 += 32) {
      v16bf a0 = load_a_frag(a0p, k0, lh);
      v16bf a1 = load_a_frag(a1p, k0, lh);
      v16bf b0 = load_b_frag(b0p, k0, lh);
      v16bf b1 = load_b_frag(b1p, k0, lh);
      acc[0][0] = wmma_bf16(a0, b0, acc[0][0]);
      acc[0][1] = wmma_bf16(a0, b1, acc[0][1]);
      acc[1][0] = wmma_bf16(a1, b0, acc[1][0]);
      acc[1][1] = wmma_bf16(a1, b1, acc[1][1]);
    }
  }
  if (A2 != nullptr) {
    const unsigned short* a0p = A2 + (long)(m0 + lr) * HID;
    const unsigned short* a1p = A2 + (long)(m0 + 16 + lr) * HID;
    const unsigned short* b0p = W2 + (long)(n0 + lr) * HID;
    const unsigned short* b1p = W2 + (long)(n0 + 16 + lr) * HID;
    for (int k0 = 0; k0 < HID; k0 += 32) {
      v16bf a0 = load_a_frag(a0p, k0, lh);
      v16bf a1 = load_a_frag(a1p, k0, lh);
      v16bf b0 = load_b_frag(b0p, k0, lh);
      v16bf b1 = load_b_frag(b1p, k0, lh);
      acc[0][0] = wmma_bf16(a0, b0, acc[0][0]);
      acc[0][1] = wmma_bf16(a0, b1, acc[0][1]);
      acc[1][0] = wmma_bf16(a1, b0, acc[1][0]);
      acc[1][1] = wmma_bf16(a1, b1, acc[1][1]);
    }
  }
#pragma unroll
  for (int mi = 0; mi < 2; ++mi)
#pragma unroll
    for (int ni = 0; ni < 2; ++ni) {
      const int col = n0 + ni * 16 + lr;
      float badd = 0.0f;
      if (bias1) badd += bias1[col];
      if (bias2) badd += bias2[col];
#pragma unroll
      for (int v = 0; v < 8; ++v) {
        const int row = m0 + mi * 16 + lh * 8 + v;
        float val = acc[mi][ni][v] + badd;
        if (addMat) val += addMat[(long)row * HID + col];
        outB[(long)row * HID + col] = f32_to_bf16(tanhf(val));
      }
    }
}

// ----------------------- logits[t] = h1 @ v_w^T + v_b -----------------------
// M = BATCH, N = VOCAB (10000, ragged), K = HID. f32 output.
// Wave tile 32x64, WG tile 64x256 -> 2 x 40 WGs per step.

__global__ void __launch_bounds__(256) k_logits(const unsigned short* __restrict__ A,
                                                const unsigned short* __restrict__ Vw,
                                                const float* __restrict__ vb,
                                                float* __restrict__ out) {
  const int wave = threadIdx.x >> 5, lane = threadIdx.x & 31;
  const int lr = lane & 15, lh = lane >> 4;
  const int m0 = blockIdx.x * 64 + (wave & 1) * 32;
  const int n0 = blockIdx.y * 256 + (wave >> 1) * 64;

  const unsigned short* ap[2];
  ap[0] = A + (long)(m0 + lr) * HID;
  ap[1] = A + (long)(m0 + 16 + lr) * HID;
  const unsigned short* bp[4];
#pragma unroll
  for (int ni = 0; ni < 4; ++ni) {
    int c = n0 + ni * 16 + lr;
    if (c >= VOCAB) c = VOCAB - 1;  // clamp: OOB cols computed but never stored
    bp[ni] = Vw + (long)c * HID;
  }

  v8f acc[2][4] = {};
  for (int k0 = 0; k0 < HID; k0 += 32) {
    v16bf a[2], b[4];
#pragma unroll
    for (int mi = 0; mi < 2; ++mi) a[mi] = load_a_frag(ap[mi], k0, lh);
#pragma unroll
    for (int ni = 0; ni < 4; ++ni) b[ni] = load_b_frag(bp[ni], k0, lh);
#pragma unroll
    for (int mi = 0; mi < 2; ++mi)
#pragma unroll
      for (int ni = 0; ni < 4; ++ni) acc[mi][ni] = wmma_bf16(a[mi], b[ni], acc[mi][ni]);
  }
#pragma unroll
  for (int mi = 0; mi < 2; ++mi)
#pragma unroll
    for (int ni = 0; ni < 4; ++ni) {
      const int col = n0 + ni * 16 + lr;
      if (col < VOCAB) {
        const float bias = vb[col];
#pragma unroll
        for (int v = 0; v < 8; ++v) {
          const int row = m0 + mi * 16 + lh * 8 + v;
          out[(long)row * VOCAB + col] = acc[mi][ni][v] + bias;
        }
      }
    }
}

// ------------------------------- host driver --------------------------------

extern "C" void kernel_launch(void* const* d_in, const int* in_sizes, int n_in,
                              void* d_out, int out_size, void* d_ws, size_t ws_size,
                              hipStream_t stream) {
  (void)in_sizes; (void)n_in; (void)out_size; (void)ws_size;

  const int*   inputs = (const int*)  d_in[0];
  const float* hidden = (const float*)d_in[1];
  const float* emb    = (const float*)d_in[2];
  const float* u0_w   = (const float*)d_in[3];
  const float* u0_b   = (const float*)d_in[4];
  const float* w0_w   = (const float*)d_in[5];
  const float* w0_b   = (const float*)d_in[6];
  const float* u1_w   = (const float*)d_in[7];
  const float* u1_b   = (const float*)d_in[8];
  const float* w1_w   = (const float*)d_in[9];
  const float* w1_b   = (const float*)d_in[10];
  const float* v_w    = (const float*)d_in[11];
  const float* v_b    = (const float*)d_in[12];
  float* out = (float*)d_out;

  char* ws = (char*)d_ws;
  size_t off = 0;
  auto alloc = [&](size_t bytes) -> void* {
    void* p = ws + off;
    off = (off + bytes + 255) & ~(size_t)255;
    return p;
  };
  unsigned short* emb16 = (unsigned short*)alloc((size_t)VOCAB * EMBD * 2);
  unsigned short* u0w16 = (unsigned short*)alloc((size_t)HID * EMBD * 2);
  unsigned short* w0w16 = (unsigned short*)alloc((size_t)HID * HID * 2);
  unsigned short* u1w16 = (unsigned short*)alloc((size_t)HID * HID * 2);
  unsigned short* w1w16 = (unsigned short*)alloc((size_t)HID * HID * 2);
  unsigned short* vw16  = (unsigned short*)alloc((size_t)VOCAB * HID * 2);
  float*          X     = (float*)alloc((size_t)SEQL * BATCH * HID * 4);
  unsigned short* h0a   = (unsigned short*)alloc((size_t)BATCH * HID * 2);
  unsigned short* h0b   = (unsigned short*)alloc((size_t)BATCH * HID * 2);
  unsigned short* h1a   = (unsigned short*)alloc((size_t)BATCH * HID * 2);
  unsigned short* h1b   = (unsigned short*)alloc((size_t)BATCH * HID * 2);

  auto cvt = [&](const float* src, unsigned short* dst, long n) {
    k_f32_to_bf16<<<dim3((unsigned)((n + 255) / 256)), dim3(256), 0, stream>>>(src, dst, (int)n);
  };
  cvt(emb,  emb16, (long)VOCAB * EMBD);
  cvt(u0_w, u0w16, (long)HID * EMBD);
  cvt(w0_w, w0w16, (long)HID * HID);
  cvt(u1_w, u1w16, (long)HID * HID);
  cvt(w1_w, w1w16, (long)HID * HID);
  cvt(v_w,  vw16,  (long)VOCAB * HID);
  cvt(hidden,               h0a, (long)BATCH * HID);
  cvt(hidden + BATCH * HID, h1a, (long)BATCH * HID);

  // Embedding projection for all 64 steps in one parallel GEMM (8192x1024x1024).
  k_x_gemm<<<dim3(SEQL * BATCH / 64, HID / 256), dim3(256), 0, stream>>>(
      inputs, emb16, u0w16, u0_b, w0_b, X);

  unsigned short *h0c = h0a, *h0n = h0b, *h1c = h1a, *h1n = h1b;
  const dim3 gHid(BATCH / 64, HID / 128);
  const dim3 gLog(BATCH / 64, (VOCAB + 255) / 256);
  for (int t = 0; t < SEQL; ++t) {
    k_rnn_layer<<<gHid, dim3(256), 0, stream>>>(h0c, w0w16, nullptr, nullptr,
                                                X + (size_t)t * BATCH * HID,
                                                nullptr, nullptr, h0n);
    k_rnn_layer<<<gHid, dim3(256), 0, stream>>>(h1c, w1w16, h0n, u1w16,
                                                nullptr, w1_b, u1_b, h1n);
    k_logits<<<gLog, dim3(256), 0, stream>>>(h1n, vw16, v_b,
                                             out + (size_t)t * BATCH * VOCAB);
    unsigned short* tmp;
    tmp = h0c; h0c = h0n; h0n = tmp;
    tmp = h1c; h1c = h1n; h1n = tmp;
  }

  // Final hidden state [2, B, H] appended after logits, in f32.
  float* outh = out + (size_t)SEQL * BATCH * VOCAB;
  k_bf16_to_f32<<<dim3((BATCH * HID + 255) / 256), dim3(256), 0, stream>>>(h0c, outh,
                                                                           BATCH * HID);
  k_bf16_to_f32<<<dim3((BATCH * HID + 255) / 256), dim3(256), 0, stream>>>(h1c,
                                                                           outh + BATCH * HID,
                                                                           BATCH * HID);
}
